// Decoder_64003602645549
// MI455X (gfx1250) — matrix operand order repaired and checked by reading, not compiled, not run on hardware
//
#include <hip/hip_runtime.h>
#include <hip/hip_bf16.h>
#include <math.h>

typedef __attribute__((ext_vector_type(16))) __bf16 v16bf;
typedef __attribute__((ext_vector_type(8)))  __bf16 v8bf;
typedef __attribute__((ext_vector_type(8)))  float  v8f;

#define BATCH 32
#define SEQL  2048
#define HID   1024
#define COLS  64          // columns (b,l pairs) per workgroup
#define NSUB  4           // COLS/16 N-subtiles per wave
#define LDS_STRIDE 1048   // bf16 elements; 524 dwords -> conflict-free b128 B-loads

static __device__ __forceinline__ __bf16 f2bf(float f) {
    unsigned u = __float_as_uint(f);
    unsigned r = (u + 0x7fffu + ((u >> 16) & 1u)) >> 16;   // RNE
    return __builtin_bit_cast(__bf16, (unsigned short)r);
}

// ---------------- fp32 -> bf16 weight conversion ----------------
__global__ void f32_to_bf16_kernel(const float* __restrict__ src, __bf16* __restrict__ dst, int n) {
    int i = blockIdx.x * 256 + threadIdx.x;
    if (i < n) dst[i] = f2bf(src[i]);
}

// ---------------- small dense linear: Y[b,i] = dot(X[b,:], W[i,:]) + bias[i] ----------------
__global__ void linear_kernel(const float* __restrict__ X, const float* __restrict__ W,
                              const float* __restrict__ bias, float* __restrict__ Y) {
    int t = blockIdx.x * 256 + threadIdx.x;   // [0, B*HID)
    int b = t >> 10, i = t & 1023;
    const float* x = X + ((size_t)b << 10);
    const float* w = W + ((size_t)i << 10);
    float a0 = 0.f, a1 = 0.f, a2 = 0.f, a3 = 0.f;
    for (int k = 0; k < HID; k += 4) {
        a0 += x[k + 0] * w[k + 0];
        a1 += x[k + 1] * w[k + 1];
        a2 += x[k + 2] * w[k + 2];
        a3 += x[k + 3] * w[k + 3];
    }
    Y[t] = (a0 + a1) + (a2 + a3) + bias[i];
}

// ---------------- fused attention-score GEMM ----------------
// att[b,l] = sum_h V[h] * tanh( (W_ctx @ context[b,l])[h] + inp[b,h] + b_ctx[h] )
// One workgroup per 64 flattened (b,l) columns staged in LDS as bf16.
// 8 waves x 8 M-tiles; each A register load feeds 4 WMMAs (4 N-subtiles).
// The asm memory barrier per mt-iteration stops the compiler from hoisting
// all LDS B-operand loads out of the loop (which previously spilled the
// whole context tile to scratch and serialized every WMMA on a reload).
__global__ void __launch_bounds__(256)
att_kernel(const __bf16* __restrict__ Wb, const float* __restrict__ context,
           const float* __restrict__ inp, const float* __restrict__ bctx,
           const float* __restrict__ V, float* __restrict__ att) {
    __shared__ __bf16 ctxLds[COLS * LDS_STRIDE];   // 64 cols x 1024 K (bf16, padded)
    __shared__ float  biasLds[HID];                // inp[b,h] + b_ctx[h]
    __shared__ float  vLds[HID];
    __shared__ float  attLds[COLS];

    const int tid  = threadIdx.x;
    const int lane = tid & 31;
    const int wave = tid >> 5;
    const int bb   = blockIdx.x >> 5;              // 32 blocks of 64 cols per batch row

    // stage context tile (fp32 -> bf16) into LDS, coalesced
    const float* ctxBase = context + (size_t)blockIdx.x * COLS * HID;
    for (int i = tid; i < COLS * HID; i += 256) {
        int col = i >> 10, h = i & 1023;
        ctxLds[col * LDS_STRIDE + h] = f2bf(ctxBase[i]);
    }
    const float* inpB = inp + ((size_t)bb << 10);
    for (int i = tid; i < HID; i += 256) {
        biasLds[i] = inpB[i] + bctx[i];
        vLds[i]    = V[i];
    }
    if (tid < COLS) attLds[tid] = 0.f;
    __syncthreads();

    const int hiHalf = lane >> 4;                  // 0: lanes 0-15, 1: lanes 16-31
    const int koffA  = hiHalf ? 16 : 0;            // A-operand K half
    const int koffB  = hiHalf ? 16 : 0;            // B-operand K half
    const int n      = lane & 15;                  // column within 16-wide subtile

    const __bf16* bBase0 = &ctxLds[(0 * 16 + n) * LDS_STRIDE + koffB];
    const __bf16* bBase1 = &ctxLds[(1 * 16 + n) * LDS_STRIDE + koffB];
    const __bf16* bBase2 = &ctxLds[(2 * 16 + n) * LDS_STRIDE + koffB];
    const __bf16* bBase3 = &ctxLds[(3 * 16 + n) * LDS_STRIDE + koffB];

    float colsum[NSUB] = {0.f, 0.f, 0.f, 0.f};
#pragma unroll 1
    for (int mt = wave * 8; mt < wave * 8 + 8; ++mt) {
        // keep LDS B-loads inside this loop (no LICM -> no scratch spills)
        asm volatile("" ::: "memory");
        const __bf16* Arow = Wb + ((size_t)(mt * 16 + n) << 10);
        v8f c0 = {}, c1 = {}, c2 = {}, c3 = {};
        for (int k0 = 0; k0 < HID; k0 += 32) {
            // A: 16x32 bf16 tile of W_ctx; lane holds row m, K = k0+koffA..+15
            v8bf alo = *(const v8bf*)(Arow + k0 + koffA);
            v8bf ahi = *(const v8bf*)(Arow + k0 + koffA + 8);
            v16bf a = __builtin_shufflevector(alo, ahi, 0,1,2,3,4,5,6,7,8,9,10,11,12,13,14,15);
            // B: 32x16 bf16 column tiles; one A load feeds 4 WMMAs
            v16bf bm0 = *(const v16bf*)(bBase0 + k0);
            c0 = __builtin_amdgcn_wmma_f32_16x16x32_bf16(false, a, false, bm0, (short)0, c0, false, false);
            v16bf bm1 = *(const v16bf*)(bBase1 + k0);
            c1 = __builtin_amdgcn_wmma_f32_16x16x32_bf16(false, a, false, bm1, (short)0, c1, false, false);
            v16bf bm2 = *(const v16bf*)(bBase2 + k0);
            c2 = __builtin_amdgcn_wmma_f32_16x16x32_bf16(false, a, false, bm2, (short)0, c2, false, false);
            v16bf bm3 = *(const v16bf*)(bBase3 + k0);
            c3 = __builtin_amdgcn_wmma_f32_16x16x32_bf16(false, a, false, bm3, (short)0, c3, false, false);
        }
        // epilogue: tanh + V-weighted row reduction for each N-subtile
        const int rowoff = mt * 16 + (hiHalf ? 8 : 0);
#pragma unroll
        for (int r = 0; r < 8; ++r) {
            int h = rowoff + r;
            float bias = biasLds[h], vv = vLds[h];
            colsum[0] += tanhf(c0[r] + bias) * vv;
            colsum[1] += tanhf(c1[r] + bias) * vv;
            colsum[2] += tanhf(c2[r] + bias) * vv;
            colsum[3] += tanhf(c3[r] + bias) * vv;
        }
    }
    // lanes n and n+16 hold the same output columns
#pragma unroll
    for (int j = 0; j < NSUB; ++j) {
        float s = colsum[j] + __shfl_xor(colsum[j], 16);
        if (lane < 16) atomicAdd(&attLds[j * 16 + lane], s);
    }
    __syncthreads();
    if (tid < COLS) att[(size_t)blockIdx.x * COLS + tid] = attLds[tid];
}

// ---------------- masked softmax + argmax, per batch row ----------------
__global__ void softmax_kernel(const float* __restrict__ att, const int* __restrict__ thresh,
                               float* __restrict__ alpha_raw, float* __restrict__ d_out,
                               int pass, int* __restrict__ idx_out) {
    __shared__ float red[256];
    __shared__ int   redi[256];
    __shared__ float s_max, s_sum;
    __shared__ int   s_idx;
    const int b = blockIdx.x, tid = threadIdx.x;
    const int T = thresh[b];
    const float* a = att + (size_t)b * SEQL;

    float vmax = -INFINITY; int vidx = SEQL;
    for (int l = tid; l < SEQL; l += 256) {
        float v = (l < T) ? -INFINITY : a[l];
        if (v > vmax) { vmax = v; vidx = l; }
    }
    red[tid] = vmax; redi[tid] = vidx; __syncthreads();
    for (int s = 128; s > 0; s >>= 1) {
        if (tid < s) {
            float o = red[tid + s]; int oi = redi[tid + s];
            if (o > red[tid] || (o == red[tid] && oi < redi[tid])) { red[tid] = o; redi[tid] = oi; }
        }
        __syncthreads();
    }
    if (tid == 0) { s_max = red[0]; s_idx = redi[0]; }
    __syncthreads();

    float psum = 0.f;
    for (int l = tid; l < SEQL; l += 256) {
        float e = (l < T) ? 0.f : expf(a[l] - s_max);
        alpha_raw[(size_t)b * SEQL + l] = e;
        psum += e;
    }
    red[tid] = psum; __syncthreads();
    for (int s = 128; s > 0; s >>= 1) {
        if (tid < s) red[tid] += red[tid + s];
        __syncthreads();
    }
    if (tid == 0) s_sum = red[0];
    __syncthreads();

    const float inv = 1.f / s_sum;
    float* outRow = d_out + (size_t)b * 2 * SEQL + (size_t)pass * SEQL;
    for (int l = tid; l < SEQL; l += 256) {
        float al = alpha_raw[(size_t)b * SEQL + l] * inv;
        alpha_raw[(size_t)b * SEQL + l] = al;            // raw alpha for weighted sum
        outRow[l] = al + 1e-6f;                           // alpha + EPS to output
    }
    if (tid == 0) idx_out[b] = s_idx;
}

// ---------------- weighted context sum: wctx[b,h] = sum_l alpha[b,l]*context[b,l,h] ----------------
__global__ void wsum_kernel(const float* __restrict__ alpha, const float* __restrict__ context,
                            float* __restrict__ wctx) {
    const int b = blockIdx.x >> 2;
    const int h = ((blockIdx.x & 3) << 8) + threadIdx.x;
    const float* cb = context + (size_t)b * SEQL * HID + h;
    const float* ab = alpha + (size_t)b * SEQL;
    float acc = 0.f;
    for (int l = 0; l < SEQL; ++l) acc += ab[l] * cb[(size_t)l * HID];
    wctx[((size_t)b << 10) + h] = acc;
}

// ---------------- final score: relu(concat(c_start,c_end,c_g)) . W_out + b_out ----------------
__global__ void final_kernel(const float* __restrict__ context, const int* __restrict__ sidx,
                             const int* __restrict__ eidx, const float* __restrict__ W_out,
                             const float* __restrict__ b_out, float* __restrict__ out) {
    __shared__ float red[256];
    const int b = blockIdx.x, tid = threadIdx.x;
    const float* cb = context + (size_t)b * SEQL * HID;
    const int s = sidx[b], e = eidx[b];
    float acc = 0.f;
    for (int h = tid; h < HID; h += 256) {
        acc += fmaxf(cb[(size_t)s * HID + h], 0.f) * W_out[h]
             + fmaxf(cb[(size_t)e * HID + h], 0.f) * W_out[HID + h]
             + fmaxf(cb[(size_t)(SEQL - 1) * HID + h], 0.f) * W_out[2 * HID + h];
    }
    red[tid] = acc; __syncthreads();
    for (int st = 128; st > 0; st >>= 1) {
        if (tid < st) red[tid] += red[tid + st];
        __syncthreads();
    }
    if (tid == 0) out[b] = red[0] + b_out[0];
}

extern "C" void kernel_launch(void* const* d_in, const int* in_sizes, int n_in,
                              void* d_out, int out_size, void* d_ws, size_t ws_size,
                              hipStream_t stream) {
    (void)in_sizes; (void)n_in; (void)out_size; (void)ws_size;
    const float* dec_in  = (const float*)d_in[0];
    const float* context = (const float*)d_in[1];
    const int*   offsets = (const int*)  d_in[2];
    const float* W_in_s  = (const float*)d_in[3];
    const float* b_in_s  = (const float*)d_in[4];
    const float* W_ctx_s = (const float*)d_in[5];
    const float* b_ctx_s = (const float*)d_in[6];
    const float* V_s     = (const float*)d_in[7];
    const float* W_in_e  = (const float*)d_in[8];
    const float* b_in_e  = (const float*)d_in[9];
    const float* W_ctx_e = (const float*)d_in[10];
    const float* b_ctx_e = (const float*)d_in[11];
    const float* V_e     = (const float*)d_in[12];
    const float* W_out   = (const float*)d_in[13];
    const float* b_out   = (const float*)d_in[14];
    float* dout = (float*)d_out;

    char* ws = (char*)d_ws;
    __bf16* wbs   = (__bf16*)(ws + 0);                 // 2 MB
    __bf16* wbe   = (__bf16*)(ws + (2u << 20));        // 2 MB
    float*  inp   = (float*)(ws + 4194304);            // 128 KB
    float*  att   = (float*)(ws + 4325376);            // 256 KB
    float*  alpha = (float*)(ws + 4587520);            // 256 KB
    float*  wctx  = (float*)(ws + 4849664);            // 128 KB
    float*  hid   = (float*)(ws + 4980736);            // 128 KB
    int*    sidx  = (int*)  (ws + 5111808);
    int*    eidx  = (int*)  (ws + 5111936);

    const int NWW = HID * HID;                         // 1M weight elements
    f32_to_bf16_kernel<<<(NWW + 255) / 256, 256, 0, stream>>>(W_ctx_s, wbs, NWW);
    f32_to_bf16_kernel<<<(NWW + 255) / 256, 256, 0, stream>>>(W_ctx_e, wbe, NWW);

    // ---- pass 1 (start) ----
    linear_kernel<<<BATCH * HID / 256, 256, 0, stream>>>(dec_in, W_in_s, b_in_s, inp);
    att_kernel<<<BATCH * SEQL / COLS, 256, 0, stream>>>(wbs, context, inp, b_ctx_s, V_s, att);
    softmax_kernel<<<BATCH, 256, 0, stream>>>(att, offsets, alpha, dout, 0, sidx);
    wsum_kernel<<<BATCH * 4, 256, 0, stream>>>(alpha, context, wctx);
    linear_kernel<<<BATCH * HID / 256, 256, 0, stream>>>(wctx, W_ctx_s, b_ctx_s, hid); // start_vec

    // ---- pass 2 (end) ----
    linear_kernel<<<BATCH * HID / 256, 256, 0, stream>>>(hid, W_in_e, b_in_e, inp);
    att_kernel<<<BATCH * SEQL / COLS, 256, 0, stream>>>(wbe, context, inp, b_ctx_e, V_e, att);
    softmax_kernel<<<BATCH, 256, 0, stream>>>(att, sidx, alpha, dout, 1, eidx);

    // ---- final score ----
    final_kernel<<<BATCH, 256, 0, stream>>>(context, sidx, eidx, W_out, b_out,
                                            dout + (size_t)BATCH * 2 * SEQL);
}
